// SemlaLayer_257698038064
// MI455X (gfx1250) — matrix operand order repaired and checked by reading, not compiled, not run on hardware
//
#include <hip/hip_runtime.h>
#include <hip/hip_bf16.h>
#include <math.h>

typedef __attribute__((ext_vector_type(16))) _Float16 v16h;
typedef __attribute__((ext_vector_type(8)))  float    v8f;

namespace {
constexpr int   kB  = 8;
constexpr int   kN  = 512;
constexpr int   kD  = 64;   // d_edge
constexpr int   kK  = 32;   // num_rbf == WMMA K
constexpr float kCut = 5.0f;
constexpr int   kTilesPerWave = 8;   // 8 row-tiles of 16 -> 128 pairs per wave
}

// edge[b,i,j,d] = ( sum_k rbf[b,i,j,k] * W[k,d] + bias[d] ) * mask[b,i]*mask[b,j]
__global__ __launch_bounds__(256) void semla_edge_rbf_wmma(
    const float* __restrict__ coords,          // [B,N,3] f32
    const unsigned char* __restrict__ mask,    // [B,N] bool bytes
    const float* __restrict__ W,               // [K,D] f32
    const float* __restrict__ bias,            // [D] f32
    float* __restrict__ out)                   // [B,N,N,D] f32
{
  const int lane = threadIdx.x & 31;
  const int wave = (blockIdx.x * blockDim.x + threadIdx.x) >> 5;

  // This wave covers 128 consecutive flat pair indices: fixed (b,i), j in [jbase, jbase+128)
  const long long p0 = (long long)wave * (16LL * kTilesPerWave);
  const int jbase = (int)(p0 & (kN - 1));
  const int i     = (int)((p0 >> 9) & (kN - 1));
  const int b     = (int)(p0 >> 18);

  const int  col = lane & 15;
  const bool hi  = lane >= 16;

  // ---- Load W into WMMA B-matrix layout (32x16 f16 per 16-col tile), plus bias ----
  // B k order chosen here (KxN, contiguous, same as A): lanes 0-15 hold K=0..15, lanes 16-31 hold K=16..31; N = lane&15.
  v16h Bm[4];
  float bv[4];
  const int kofsB = hi ? 16 : 0;
#pragma unroll
  for (int t = 0; t < 4; ++t) {
    const int c = t * 16 + col;
#pragma unroll
    for (int h = 0; h < 16; ++h)
      Bm[t][h] = (_Float16)W[(kofsB + h) * kD + c];
    bv[t] = bias[c];
  }

  const float* xi = coords + (size_t)(b * kN + i) * 3;
  const float xix = xi[0], xiy = xi[1], xiz = xi[2];
  const float mi = mask[(size_t)b * kN + i] ? 1.0f : 0.0f;

  constexpr float step  = kCut / (float)(kK - 1);
  constexpr float coeff = -0.5f / (step * step);
  // A layout (16-bit, 16x32): low lanes fill K 0..15; high lanes K 16..31 (matches B).
  const int base0 = hi ? 16 : 0;
  const int base1 = hi ? 24 : 8;

#pragma unroll 1
  for (int tt = 0; tt < kTilesPerWave; ++tt) {
    const int j0 = jbase + tt * 16;
    const int j  = j0 + col;               // this lane fills A-matrix row r = col

    const float* xj = coords + (size_t)(b * kN + j) * 3;
    const float dx = xix - xj[0];
    const float dy = xiy - xj[1];
    const float dz = xiz - xj[2];
    const float d  = sqrtf(dx * dx + dy * dy + dz * dz);

    const float cw = (d < kCut)
        ? 0.5f * (__cosf(d * (float)(M_PI / 5.0)) + 1.0f)
        : 0.0f;

    v16h a;
#pragma unroll
    for (int h = 0; h < 8; ++h) {
      const float df = d - (float)(base0 + h) * step;
      a[h] = (_Float16)(cw * __expf(coeff * df * df));
    }
#pragma unroll
    for (int h = 0; h < 8; ++h) {
      const float df = d - (float)(base1 + h) * step;
      a[8 + h] = (_Float16)(cw * __expf(coeff * df * df));
    }

    // One WMMA per 16-wide column tile of d_edge (K=32 reduced in a single instruction).
    v8f acc[4];
#pragma unroll
    for (int t = 0; t < 4; ++t) {
      v8f z = {};
      acc[t] = __builtin_amdgcn_wmma_f32_16x16x32_f16(
          /*neg_a=*/false, a, /*neg_b=*/false, Bm[t],
          /*c_mod=*/(short)0, z, /*reuse_a=*/false, /*reuse_b=*/false);
    }

    // Masks for the 8 output rows this lane stores (C/D layout: VGPR k -> M = k + 8*hi).
    const unsigned long long mj8 =
        *(const unsigned long long*)(mask + (size_t)b * kN + j0 + (hi ? 8 : 0));
    const size_t obase = ((size_t)(b * kN + i) * kN + j0) * (size_t)kD;

#pragma unroll
    for (int k = 0; k < 8; ++k) {
      const int M = (hi ? 8 : 0) + k;
      const float m = (mi != 0.0f && ((mj8 >> (8 * k)) & 0xffULL)) ? 1.0f : 0.0f;
      const size_t row = obase + (size_t)M * kD + col;
#pragma unroll
      for (int t = 0; t < 4; ++t)
        out[row + (size_t)t * 16] = (acc[t][k] + bv[t]) * m;
    }
  }
}

extern "C" void kernel_launch(void* const* d_in, const int* in_sizes, int n_in,
                              void* d_out, int out_size, void* d_ws, size_t ws_size,
                              hipStream_t stream) {
  (void)in_sizes; (void)n_in; (void)out_size; (void)d_ws; (void)ws_size;
  const float*         coords = (const float*)d_in[0];
  const unsigned char* mask   = (const unsigned char*)d_in[1];
  const float*         W      = (const float*)d_in[2];
  const float*         bias   = (const float*)d_in[3];
  float*               out    = (float*)d_out;

  const long long totalPairs = (long long)kB * kN * kN;           // 2,097,152
  const long long waves      = totalPairs / (16LL * kTilesPerWave); // 16,384
  const int threadsPerBlock  = 256;                                 // 8 waves
  const int blocks           = (int)(waves / (threadsPerBlock / 32)); // 2,048

  semla_edge_rbf_wmma<<<blocks, threadsPerBlock, 0, stream>>>(coords, mask, W, bias, out);
}
